// Encoder_13907104104758
// MI455X (gfx1250) — matrix operand order repaired and checked by reading, not compiled, not run on hardware
//
#include <hip/hip_runtime.h>
#include <hip/hip_bf16.h>
#include <stdint.h>

typedef __attribute__((ext_vector_type(16))) __bf16 v16bf;
typedef __attribute__((ext_vector_type(8)))  __bf16 v8bf;
typedef __attribute__((ext_vector_type(4)))  __bf16 v4bf;
typedef __attribute__((ext_vector_type(8)))  float  v8f;
typedef int v4i_gcc __attribute__((vector_size(16)));   // matches builtin param type

#define E_   512
#define H_   1024
#define B_   256
#define T_   128
#define G4H  (4 * H_)      // 4096
#define BT   (B_ * T_)     // 32768

// GEMM tiling
#define BM   128
#define BN   128
#define BK   32
#define LDK  (BK + 8)      // padded LDS row (80B) to spread banks

#define AS1_ __attribute__((address_space(1)))
#define AS3_ __attribute__((address_space(3)))

// ---- gfx1250 async global->LDS copy (probe via __has_builtin) ----
#if defined(__gfx1250__) && __has_builtin(__builtin_amdgcn_global_load_async_to_lds_b128) && __has_builtin(__builtin_amdgcn_s_wait_asynccnt)
#define USE_ASYNC_LDS 1
#else
#define USE_ASYNC_LDS 0
#endif

__device__ __forceinline__ void async_copy16(const __bf16* g, __bf16* l) {
#if USE_ASYNC_LDS
  __builtin_amdgcn_global_load_async_to_lds_b128(
      (AS1_ v4i_gcc*)(AS1_ void*)(void*)(__bf16*)g,
      (AS3_ v4i_gcc*)(AS3_ void*)(void*)l,
      0, 0);
#else
  (void)g; (void)l;
#endif
}

__device__ __forceinline__ void wait_async_all() {
#if USE_ASYNC_LDS
  __builtin_amdgcn_s_wait_asynccnt(0);
#endif
}

// ---------------- f32 -> bf16 convert ----------------
__global__ void f32_to_bf16_kernel(const float* __restrict__ in,
                                   __bf16* __restrict__ out, int n) {
  int i = blockIdx.x * blockDim.x + threadIdx.x;
  if (i < n) out[i] = (__bf16)in[i];
}

// ---------------- bias_sum = b_ih + b_hh ----------------
__global__ void bias_sum_kernel(const float* __restrict__ a,
                                const float* __restrict__ b,
                                float* __restrict__ out, int n) {
  int i = blockIdx.x * blockDim.x + threadIdx.x;
  if (i < n) out[i] = a[i] + b[i];
}

// ---------------- embedding gather + relu -> bf16 ----------------
__global__ void embed_relu_bf16_kernel(const int* __restrict__ tokens,
                                       const float* __restrict__ emb,
                                       __bf16* __restrict__ x) {
  int row = blockIdx.x;
  int tok = tokens[row];
  const float4* src = reinterpret_cast<const float4*>(emb + (size_t)tok * E_);
  v4bf* dst = reinterpret_cast<v4bf*>(x + (size_t)row * E_);
  float4 v = src[threadIdx.x];
  v4bf o;
  o[0] = (__bf16)fmaxf(v.x, 0.0f);
  o[1] = (__bf16)fmaxf(v.y, 0.0f);
  o[2] = (__bf16)fmaxf(v.z, 0.0f);
  o[3] = (__bf16)fmaxf(v.w, 0.0f);
  dst[threadIdx.x] = o;
}

// ---- fragment load + 8 WMMAs for one staged K-slice ----
__device__ __forceinline__ void wmma_tiles(const __bf16* __restrict__ bufA,
                                           const __bf16* __restrict__ bufB,
                                           int wm0, int wn0, int hl, int lm,
                                           v8f acc[2][4]) {
  v16bf afrag[2], bfrag[4];
#pragma unroll
  for (int mt = 0; mt < 2; ++mt) {
    const __bf16* ap = bufA + (wm0 + mt * 16 + lm) * LDK + hl * 8;
    v8bf a0 = *reinterpret_cast<const v8bf*>(ap);
    v8bf a1 = *reinterpret_cast<const v8bf*>(ap + 16);
#pragma unroll
    for (int i = 0; i < 8; ++i) { afrag[mt][i] = a0[i]; afrag[mt][i + 8] = a1[i]; }
  }
#pragma unroll
  for (int nt = 0; nt < 4; ++nt) {
    const __bf16* bp = bufB + (wn0 + nt * 16 + lm) * LDK + hl * 8;
    v8bf b0 = *reinterpret_cast<const v8bf*>(bp);
    v8bf b1 = *reinterpret_cast<const v8bf*>(bp + 16);
#pragma unroll
    for (int i = 0; i < 8; ++i) { bfrag[nt][i] = b0[i]; bfrag[nt][i + 8] = b1[i]; }
  }
#pragma unroll
  for (int mt = 0; mt < 2; ++mt)
#pragma unroll
    for (int nt = 0; nt < 4; ++nt)
      acc[mt][nt] = __builtin_amdgcn_wmma_f32_16x16x32_bf16(
          false, afrag[mt], false, bfrag[nt],
          (short)0, acc[mt][nt], false, false);
}

// ---------------- bf16 WMMA GEMM: C = A[M,K] * W[N,K]^T (+Cin or +bias) ----
// Block: 256 threads (8 waves), block tile 128x128, K-step 32.
// Double-buffered LDS, staged with GLOBAL_LOAD_ASYNC_TO_LDS_B128 when available.
// Waves arranged 4(M-rows) x 2(N-cols); each wave: 2 M-tiles x 4 N-tiles.
__global__ __launch_bounds__(256)
void gemm_bf16_wmma(const __bf16* __restrict__ A, int lda,
                    const __bf16* __restrict__ W, int ldw,
                    const float* __restrict__ Cin, long long ldc_in,
                    const float* __restrict__ bias,
                    float* __restrict__ C, long long ldc,
                    int K) {
  __shared__ __bf16 lA[2][BM * LDK];
  __shared__ __bf16 lB[2][BN * LDK];

  const int tid  = threadIdx.x;
  const int lane = tid & 31;
  const int wave = tid >> 5;
  const int wr   = wave >> 1;     // 0..3 : M wave-row
  const int wc   = wave & 1;      // 0..1 : N wave-col
  const int hl   = lane >> 4;     // lane half
  const int lm   = lane & 15;

  const int blockM = blockIdx.y * BM;
  const int blockN = blockIdx.x * BN;
  const int wm0 = wr * 32;        // wave M origin in block (2 tiles)
  const int wn0 = wc * 64;        // wave N origin in block (4 tiles)

  // staging chunk coordinates (each thread: 2 x 16B chunks per tile)
  const int ch0  = tid * 2;
  const int row0 = ch0 >> 2;
  const int col0 = (ch0 & 3) * 8;
  const int ch1  = ch0 + 1;
  const int row1 = ch1 >> 2;
  const int col1 = (ch1 & 3) * 8;
  const int la0  = row0 * LDK + col0;
  const int la1  = row1 * LDK + col1;

  // ---- accumulator init ----
  v8f acc[2][4];
  if (Cin) {
#pragma unroll
    for (int mt = 0; mt < 2; ++mt)
#pragma unroll
      for (int nt = 0; nt < 4; ++nt) {
        const float* cp = Cin + (size_t)(blockM + wm0 + mt * 16 + 8 * hl) * (size_t)ldc_in
                              + blockN + wn0 + nt * 16 + lm;
#pragma unroll
        for (int r = 0; r < 8; ++r) acc[mt][nt][r] = cp[(size_t)r * (size_t)ldc_in];
      }
  } else {
#pragma unroll
    for (int mt = 0; mt < 2; ++mt)
#pragma unroll
      for (int nt = 0; nt < 4; ++nt)
#pragma unroll
        for (int r = 0; r < 8; ++r) acc[mt][nt][r] = 0.0f;
  }

  const __bf16* Ag0 = A + (size_t)(blockM + row0) * lda + col0;
  const __bf16* Ag1 = A + (size_t)(blockM + row1) * lda + col1;
  const __bf16* Wg0 = W + (size_t)(blockN + row0) * ldw + col0;
  const __bf16* Wg1 = W + (size_t)(blockN + row1) * ldw + col1;

#if USE_ASYNC_LDS
  // -------- async double-buffered pipeline --------
  const int nk = K / BK;
  // prologue: tile 0 -> buffer 0
  async_copy16(Ag0, &lA[0][la0]);
  async_copy16(Ag1, &lA[0][la1]);
  async_copy16(Wg0, &lB[0][la0]);
  async_copy16(Wg1, &lB[0][la1]);
  for (int kk = 0; kk < nk; ++kk) {
    const int buf = kk & 1;
    wait_async_all();      // my tile-kk copies are in LDS
    __syncthreads();       // everyone's are; everyone done reading buf^1
    if (kk + 1 < nk) {     // overlap tile kk+1 copy with tile kk compute
      const int k1 = (kk + 1) * BK;
      async_copy16(Ag0 + k1, &lA[buf ^ 1][la0]);
      async_copy16(Ag1 + k1, &lA[buf ^ 1][la1]);
      async_copy16(Wg0 + k1, &lB[buf ^ 1][la0]);
      async_copy16(Wg1 + k1, &lB[buf ^ 1][la1]);
      if (kk + 2 < nk) {
        __builtin_prefetch(Ag0 + k1 + BK, 0, 3);
        __builtin_prefetch(Wg0 + k1 + BK, 0, 3);
      }
    }
    wmma_tiles(&lA[buf][0], &lB[buf][0], wm0, wn0, hl, lm, acc);
  }
#else
  // -------- fallback: single buffer, batched register staging --------
  for (int k0 = 0; k0 < K; k0 += BK) {
    v8bf t0 = *reinterpret_cast<const v8bf*>(Ag0 + k0);
    v8bf t1 = *reinterpret_cast<const v8bf*>(Ag1 + k0);
    v8bf t2 = *reinterpret_cast<const v8bf*>(Wg0 + k0);
    v8bf t3 = *reinterpret_cast<const v8bf*>(Wg1 + k0);
    *reinterpret_cast<v8bf*>(&lA[0][la0]) = t0;
    *reinterpret_cast<v8bf*>(&lA[0][la1]) = t1;
    *reinterpret_cast<v8bf*>(&lB[0][la0]) = t2;
    *reinterpret_cast<v8bf*>(&lB[0][la1]) = t3;
    if (k0 + BK < K) {
      __builtin_prefetch(Ag0 + k0 + BK, 0, 3);
      __builtin_prefetch(Wg0 + k0 + BK, 0, 3);
    }
    __syncthreads();
    wmma_tiles(&lA[0][0], &lB[0][0], wm0, wn0, hl, lm, acc);
    __syncthreads();
  }
#endif

  // ---- epilogue: bias + store ----
#pragma unroll
  for (int nt = 0; nt < 4; ++nt) {
    const float bv = bias ? bias[blockN + wn0 + nt * 16 + lm] : 0.0f;
#pragma unroll
    for (int mt = 0; mt < 2; ++mt) {
      float* cp = C + (size_t)(blockM + wm0 + mt * 16 + 8 * hl) * (size_t)ldc
                    + blockN + wn0 + nt * 16 + lm;
#pragma unroll
      for (int r = 0; r < 8; ++r) cp[(size_t)r * (size_t)ldc] = acc[mt][nt][r] + bv;
    }
  }
}

// ---------------- LSTM pointwise cell update ----------------
__device__ __forceinline__ float sigf(float x) { return 1.0f / (1.0f + expf(-x)); }

__global__ void lstm_cell_kernel(const float* __restrict__ gates,  // [B, 4H]
                                 float* __restrict__ h,            // [B, H]
                                 float* __restrict__ c,            // [B, H]
                                 __bf16* __restrict__ h_bf,        // [B, H]
                                 float* __restrict__ y_f32,        // [B,T,H] or null
                                 __bf16* __restrict__ y_bf,        // [B,T,H] or null
                                 int t) {
  int idx = blockIdx.x * blockDim.x + threadIdx.x;  // 0 .. B*H-1
  int b = idx >> 10;          // / H_
  int j = idx & (H_ - 1);
  const float* g = gates + (size_t)b * G4H;
  float ig = sigf(g[j]);
  float fg = sigf(g[j + H_]);
  float gc = tanhf(g[j + 2 * H_]);
  float og = sigf(g[j + 3 * H_]);
  float cc = fg * c[idx] + ig * gc;
  float hh = og * tanhf(cc);
  c[idx] = cc;
  h[idx] = hh;
  h_bf[idx] = (__bf16)hh;
  size_t yoff = (size_t)b * T_ * H_ + (size_t)t * H_ + j;
  if (y_f32) y_f32[yoff] = hh;
  if (y_bf)  y_bf[yoff] = (__bf16)hh;
}

extern "C" void kernel_launch(void* const* d_in, const int* in_sizes, int n_in,
                              void* d_out, int out_size, void* d_ws, size_t ws_size,
                              hipStream_t stream) {
  (void)in_sizes; (void)n_in; (void)out_size; (void)ws_size;
  const int*   tokens = (const int*)d_in[0];
  const float* emb    = (const float*)d_in[1];
  const float* Wih0   = (const float*)d_in[2];
  const float* Whh0   = (const float*)d_in[3];
  const float* bih0   = (const float*)d_in[4];
  const float* bhh0   = (const float*)d_in[5];
  const float* Wih1   = (const float*)d_in[6];
  const float* Whh1   = (const float*)d_in[7];
  const float* bih1   = (const float*)d_in[8];
  const float* bhh1   = (const float*)d_in[9];

  float* out = (float*)d_out;
  float* y1  = out;                                   // [B, T, H]
  float* hn  = out + (size_t)BT * H_;                 // [2, B, H]
  float* cn  = hn + 2 * (size_t)B_ * H_;              // [2, B, H]

  // -------- workspace carve --------
  uint8_t* p = (uint8_t*)d_ws;
  auto alloc = [&](size_t bytes) -> void* {
    void* q = (void*)p;
    p += (bytes + 255) & ~(size_t)255;
    return q;
  };
  float*  xg     = (float*) alloc((size_t)BT * G4H * 4);   // 512 MB, reused per layer
  __bf16* x0b    = (__bf16*)alloc((size_t)BT * E_ * 2);    // layer0 input
  __bf16* y0b    = (__bf16*)alloc((size_t)BT * H_ * 2);    // layer0 output (bf16)
  __bf16* wih0b  = (__bf16*)alloc((size_t)G4H * E_ * 2);
  __bf16* whh0b  = (__bf16*)alloc((size_t)G4H * H_ * 2);
  __bf16* wih1b  = (__bf16*)alloc((size_t)G4H * H_ * 2);
  __bf16* whh1b  = (__bf16*)alloc((size_t)G4H * H_ * 2);
  float*  bias0  = (float*) alloc((size_t)G4H * 4);
  float*  bias1  = (float*) alloc((size_t)G4H * 4);
  float*  gates  = (float*) alloc((size_t)B_ * G4H * 4);
  float*  h0     = (float*) alloc((size_t)B_ * H_ * 4);
  float*  c0     = (float*) alloc((size_t)B_ * H_ * 4);
  float*  h1     = (float*) alloc((size_t)B_ * H_ * 4);
  float*  c1     = (float*) alloc((size_t)B_ * H_ * 4);
  __bf16* hbf    = (__bf16*)alloc((size_t)B_ * H_ * 2);

  // -------- weight / bias preprocessing --------
  {
    int n;
    n = G4H * E_;
    f32_to_bf16_kernel<<<(n + 255) / 256, 256, 0, stream>>>(Wih0, wih0b, n);
    n = G4H * H_;
    f32_to_bf16_kernel<<<(n + 255) / 256, 256, 0, stream>>>(Whh0, whh0b, n);
    f32_to_bf16_kernel<<<(n + 255) / 256, 256, 0, stream>>>(Wih1, wih1b, n);
    f32_to_bf16_kernel<<<(n + 255) / 256, 256, 0, stream>>>(Whh1, whh1b, n);
    bias_sum_kernel<<<(G4H + 255) / 256, 256, 0, stream>>>(bih0, bhh0, bias0, G4H);
    bias_sum_kernel<<<(G4H + 255) / 256, 256, 0, stream>>>(bih1, bhh1, bias1, G4H);
  }

  // -------- embed + relu --------
  embed_relu_bf16_kernel<<<BT, 128, 0, stream>>>(tokens, emb, x0b);

  const dim3 gridXG(G4H / BN, BT / BM);   // (32, 256)
  const dim3 gridG (G4H / BN, B_ / BM);   // (32, 2)
  const int cellBlocks = (B_ * H_) / 256; // 1024

  // ================= layer 0 =================
  gemm_bf16_wmma<<<gridXG, 256, 0, stream>>>(x0b, E_, wih0b, E_,
                                             nullptr, 0, bias0,
                                             xg, G4H, E_);
  (void)hipMemsetAsync(h0, 0, (size_t)B_ * H_ * 4, stream);
  (void)hipMemsetAsync(c0, 0, (size_t)B_ * H_ * 4, stream);
  (void)hipMemsetAsync(hbf, 0, (size_t)B_ * H_ * 2, stream);
  for (int t = 0; t < T_; ++t) {
    gemm_bf16_wmma<<<gridG, 256, 0, stream>>>(hbf, H_, whh0b, H_,
                                              xg + (size_t)t * G4H, (long long)T_ * G4H,
                                              nullptr, gates, G4H, H_);
    lstm_cell_kernel<<<cellBlocks, 256, 0, stream>>>(gates, h0, c0, hbf,
                                                     nullptr, y0b, t);
  }

  // ================= layer 1 =================
  gemm_bf16_wmma<<<gridXG, 256, 0, stream>>>(y0b, H_, wih1b, H_,
                                             nullptr, 0, bias1,
                                             xg, G4H, H_);
  (void)hipMemsetAsync(h1, 0, (size_t)B_ * H_ * 4, stream);
  (void)hipMemsetAsync(c1, 0, (size_t)B_ * H_ * 4, stream);
  (void)hipMemsetAsync(hbf, 0, (size_t)B_ * H_ * 2, stream);
  for (int t = 0; t < T_; ++t) {
    gemm_bf16_wmma<<<gridG, 256, 0, stream>>>(hbf, H_, whh1b, H_,
                                              xg + (size_t)t * G4H, (long long)T_ * G4H,
                                              nullptr, gates, G4H, H_);
    lstm_cell_kernel<<<cellBlocks, 256, 0, stream>>>(gates, h1, c1, hbf,
                                                     y1, nullptr, t);
  }

  // -------- final states --------
  const size_t st = (size_t)B_ * H_ * 4;
  (void)hipMemcpyAsync(hn,            h0, st, hipMemcpyDeviceToDevice, stream);
  (void)hipMemcpyAsync(hn + B_ * H_,  h1, st, hipMemcpyDeviceToDevice, stream);
  (void)hipMemcpyAsync(cn,            c0, st, hipMemcpyDeviceToDevice, stream);
  (void)hipMemcpyAsync(cn + B_ * H_,  c1, st, hipMemcpyDeviceToDevice, stream);
}
